// CFNCell_61692910239821
// MI455X (gfx1250) — matrix operand order repaired and checked by reading, not compiled, not run on hardware
//
#include <hip/hip_runtime.h>

#define B_DIM 4096
#define DIN   2048
#define H_DIM 2048

typedef __attribute__((ext_vector_type(16))) __bf16 v16bf;
typedef __attribute__((ext_vector_type(8)))  __bf16 v8bf;
typedef __attribute__((ext_vector_type(8)))  float  v8f;
typedef __attribute__((ext_vector_type(4)))  unsigned int u32x4;
typedef __attribute__((ext_vector_type(4)))  int i32x4;
typedef __attribute__((ext_vector_type(8)))  int i32x8;

#if defined(__AMDGCN__) && __has_builtin(__builtin_amdgcn_tensor_load_to_lds)
#define USE_TDM 1
#else
#define USE_TDM 0
#endif

__device__ __forceinline__ float sigmoidf_(float x) {
  return 1.0f / (1.0f + __expf(-x));
}

__device__ __forceinline__ void wait_tensor0() {
#if defined(__AMDGCN__) && __has_builtin(__builtin_amdgcn_s_wait_tensorcnt)
  __builtin_amdgcn_s_wait_tensorcnt(0);
#elif defined(__AMDGCN__)
  asm volatile("s_wait_tensorcnt 0x0" ::: "memory");
#endif
}

// ---------------------------------------------------------------------------
// Phase 1: pack f32 -> bf16 (one-time; GEMM loop then has zero conversion work
// and half the staging bytes; whole bf16 working set ~72MB lives in 192MB L2).
// ---------------------------------------------------------------------------
__global__ __launch_bounds__(256)
void cvt_bf16_kernel(const float* __restrict__ src, __bf16* __restrict__ dst, int n) {
  const int i = (blockIdx.x * 256 + threadIdx.x) * 8;
  if (i < n) {
    const float4 a = *(const float4*)(src + i);
    const float4 b = *(const float4*)(src + i + 4);
    union { __bf16 h[8]; uint4 u; } p;
    p.h[0] = (__bf16)a.x; p.h[1] = (__bf16)a.y; p.h[2] = (__bf16)a.z; p.h[3] = (__bf16)a.w;
    p.h[4] = (__bf16)b.x; p.h[5] = (__bf16)b.y; p.h[6] = (__bf16)b.z; p.h[7] = (__bf16)b.w;
    *(uint4*)(dst + i) = p.u;
  }
}

// ---------------------------------------------------------------------------
// TDM descriptor issue: 2D bf16 tile (tileRows x tileCols), row stride in
// elements, DMA'd global -> LDS. D# layout per CDNA5 ISA ch.8.
// ---------------------------------------------------------------------------
#if USE_TDM
__device__ __forceinline__ void tdm_load_tile(unsigned ldsByteOff, const __bf16* g,
                                              unsigned tileCols, unsigned tileRows,
                                              unsigned strideElems) {
  const unsigned long long ga = (unsigned long long)(size_t)g;
  u32x4 g0;
  g0[0] = 1u;                                   // count=1, user descriptor
  g0[1] = ldsByteOff;                           // lds_addr
  g0[2] = (unsigned)ga;                         // global_addr[31:0]
  g0[3] = (unsigned)(ga >> 32) | 0x80000000u;   // global_addr[56:32] | type=2
  i32x8 g1;
  g1[0] = 0x00010000;                           // data_size=1 (2 bytes)
  g1[1] = 0;                                    // no atomic barrier; tensor_dim0[15:0]=0
  g1[2] = 0x00004000;                           // tensor_dim0=2^30 (no OOB); td1 lo=0
  g1[3] = (int)(0x00004000u | (tileCols << 16));// tensor_dim1=2^30; tile_dim0
  g1[4] = (int)tileRows;                        // tile_dim1; tile_dim2=0
  g1[5] = (int)strideElems;                     // tensor_dim0_stride[31:0]
  g1[6] = 0;
  g1[7] = 0;
  i32x4 gz = {0, 0, 0, 0};
#if defined(__clang_major__) && (__clang_major__ >= 23)
  i32x8 z8 = {0, 0, 0, 0, 0, 0, 0, 0};
  __builtin_amdgcn_tensor_load_to_lds(g0, g1, gz, gz, z8, 0);
#else
  __builtin_amdgcn_tensor_load_to_lds(g0, g1, gz, gz, 0);
#endif
}
#endif

// ---------------------------------------------------------------------------
// Fragment loads per ISA 7.12.2 layouts.
// ---------------------------------------------------------------------------
__device__ __forceinline__ v16bf frag_a(const __bf16* s /*Mx32*/, int mBase, int lane) {
  const int m  = lane & 15;
  const int kh = lane >> 4;
  const __bf16* p = s + (mBase + m) * 32 + kh * 8;
  v8bf lo = *(const v8bf*)p;
  v8bf hi = *(const v8bf*)(p + 16);
  v16bf r;
#pragma unroll
  for (int i = 0; i < 8; ++i) { r[i] = lo[i]; r[i + 8] = hi[i]; }
  return r;
}

__device__ __forceinline__ v16bf frag_b(const __bf16* s /*32x64*/, int nBase, int lane) {
  const int n  = nBase + (lane & 15);
  const int kh = lane >> 4;
  v16bf r;
#pragma unroll
  for (int i = 0; i < 8; ++i) { r[i] = s[(kh * 8 + i) * 64 + n]; r[i + 8] = s[(16 + kh * 8 + i) * 64 + n]; }
  return r;
}

#define WMMA_BF16(A, Bm, C) \
  __builtin_amdgcn_wmma_f32_16x16x32_bf16(false, (A), false, (Bm), (short)0, (C), false, false)

// Per-buffer element offsets inside LDS staging buffer (18432 bf16 = 36864 B).
#define OFF_AS  0
#define OFF_AI  4096
#define OFF_BTU 8192
#define OFF_BTW 10240
#define OFF_BEU 12288
#define OFF_BEW 14336
#define OFF_BWX 16384
#define BUF_ELEMS 18432

// All fragment LDS loads issued up-front, then all 20 WMMAs: lets the
// scheduler use partial s_wait_dscnt instead of gating every WMMA on 0.
__device__ __forceinline__ void compute_step(const __bf16* buf, int waveM, int waveN, int lane,
                                             v8f (&accT)[2][2], v8f (&accE)[2][2],
                                             v8f (&accX)[2][2]) {
  const v16bf aS0 = frag_a(buf + OFF_AS, waveM * 32 + 0,  lane);
  const v16bf aS1 = frag_a(buf + OFF_AS, waveM * 32 + 16, lane);
  const v16bf aI0 = frag_a(buf + OFF_AI, waveM * 32 + 0,  lane);
  const v16bf aI1 = frag_a(buf + OFF_AI, waveM * 32 + 16, lane);

  v16bf bTu[2], bTw[2], bEu[2], bEw[2], bWx[2];
#pragma unroll
  for (int ni = 0; ni < 2; ++ni) {
    const int nb = waveN * 32 + ni * 16;
    bTu[ni] = frag_b(buf + OFF_BTU, nb, lane);
    bTw[ni] = frag_b(buf + OFF_BTW, nb, lane);
    bEu[ni] = frag_b(buf + OFF_BEU, nb, lane);
    bEw[ni] = frag_b(buf + OFF_BEW, nb, lane);
    bWx[ni] = frag_b(buf + OFF_BWX, nb, lane);
  }

#pragma unroll
  for (int ni = 0; ni < 2; ++ni) {
    accT[0][ni] = WMMA_BF16(aS0, bTu[ni], accT[0][ni]);
    accT[1][ni] = WMMA_BF16(aS1, bTu[ni], accT[1][ni]);
    accT[0][ni] = WMMA_BF16(aI0, bTw[ni], accT[0][ni]);
    accT[1][ni] = WMMA_BF16(aI1, bTw[ni], accT[1][ni]);

    accE[0][ni] = WMMA_BF16(aS0, bEu[ni], accE[0][ni]);
    accE[1][ni] = WMMA_BF16(aS1, bEu[ni], accE[1][ni]);
    accE[0][ni] = WMMA_BF16(aI0, bEw[ni], accE[0][ni]);
    accE[1][ni] = WMMA_BF16(aI1, bEw[ni], accE[1][ni]);

    accX[0][ni] = WMMA_BF16(aI0, bWx[ni], accX[0][ni]);
    accX[1][ni] = WMMA_BF16(aI1, bWx[ni], accX[1][ni]);
  }
}

#if !USE_TDM
__device__ __forceinline__ void copy_tile_bf16(const __bf16* __restrict__ g, int ldg,
                                               __bf16* __restrict__ s,
                                               int rows, int cols, int tid, int nthr) {
  const int chunks = (rows * cols) >> 3;
  for (int i = tid; i < chunks; i += nthr) {
    const int idx = i << 3;
    const int r = idx / cols;
    const int c = idx - r * cols;
    *(uint4*)(s + idx) = *(const uint4*)(g + (size_t)r * ldg + c);
  }
}
#endif

// ---------------------------------------------------------------------------
// Phase 2: TDM-fed, LDS double-buffered, WMMA GEMM with fused CFN epilogue.
// Block = 256 threads (8 waves); tile 128(M) x 64(N); 20 WMMAs / K-step.
// ---------------------------------------------------------------------------
__global__ __launch_bounds__(256)
void cfn_gemm_bf16_kernel(const __bf16* __restrict__ inputs_bf,
                          const __bf16* __restrict__ state_bf,
                          const __bf16* __restrict__ wtu, const __bf16* __restrict__ wtw,
                          const __bf16* __restrict__ weu, const __bf16* __restrict__ wew,
                          const __bf16* __restrict__ wwx,
                          const float* __restrict__ state_f32,
                          const float* __restrict__ theta_b,
                          const float* __restrict__ eta_b,
                          float* __restrict__ out) {
  __shared__ __bf16 smem[2][BUF_ELEMS];

  const int tid   = threadIdx.x;
  const int lane  = tid & 31;
  const int wave  = tid >> 5;
  const int waveM = wave & 3;
  const int waveN = wave >> 2;
  const int rowBase = blockIdx.y * 128;
  const int colBase = blockIdx.x * 64;

  v8f accT[2][2] = {};
  v8f accE[2][2] = {};
  v8f accX[2][2] = {};

#if USE_TDM
  // Wave-uniform (SGPR) predicate -> scalar s_cbranch, so waves 1..7 never
  // reach the tensor ops (TDM issues regardless of EXEC; exec-masking would
  // cause 8x duplicate DMA issue).
  const bool dmaWave = (__builtin_amdgcn_readfirstlane(tid) == 0);

  auto stage = [&](int k, int bufIdx) {
    const unsigned base = (unsigned)(size_t)(&smem[bufIdx][0]);
    tdm_load_tile(base + OFF_AS  * 2, state_bf  + (size_t)rowBase * H_DIM + k, 32, 128, H_DIM);
    tdm_load_tile(base + OFF_AI  * 2, inputs_bf + (size_t)rowBase * DIN   + k, 32, 128, DIN);
    tdm_load_tile(base + OFF_BTU * 2, wtu + (size_t)k * H_DIM + colBase, 64, 32, H_DIM);
    tdm_load_tile(base + OFF_BTW * 2, wtw + (size_t)k * H_DIM + colBase, 64, 32, H_DIM);
    tdm_load_tile(base + OFF_BEU * 2, weu + (size_t)k * H_DIM + colBase, 64, 32, H_DIM);
    tdm_load_tile(base + OFF_BEW * 2, wew + (size_t)k * H_DIM + colBase, 64, 32, H_DIM);
    tdm_load_tile(base + OFF_BWX * 2, wwx + (size_t)k * H_DIM + colBase, 64, 32, H_DIM);
  };
  if (dmaWave) stage(0, 0);            // prologue DMA into buffer 0
  int buf = 0;
  for (int k = 0; k < H_DIM; k += 32) {
    if (dmaWave) wait_tensor0();       // issuing wave: TENSORcnt==0 -> tiles landed
    __syncthreads();                   // publish LDS to all waves
    if (dmaWave && (k + 32) < H_DIM) stage(k + 32, buf ^ 1);  // overlap DMA w/ WMMA
    compute_step(&smem[buf][0], waveM, waveN, lane, accT, accE, accX);
    buf ^= 1;
  }
#else
  for (int k = 0; k < H_DIM; k += 32) {
    __syncthreads();
    copy_tile_bf16(state_bf  + (size_t)rowBase * H_DIM + k, H_DIM, &smem[0][OFF_AS], 128, 32, tid, 256);
    copy_tile_bf16(inputs_bf + (size_t)rowBase * DIN   + k, DIN,   &smem[0][OFF_AI], 128, 32, tid, 256);
    copy_tile_bf16(wtu + (size_t)k * H_DIM + colBase, H_DIM, &smem[0][OFF_BTU], 32, 64, tid, 256);
    copy_tile_bf16(wtw + (size_t)k * H_DIM + colBase, H_DIM, &smem[0][OFF_BTW], 32, 64, tid, 256);
    copy_tile_bf16(weu + (size_t)k * H_DIM + colBase, H_DIM, &smem[0][OFF_BEU], 32, 64, tid, 256);
    copy_tile_bf16(wew + (size_t)k * H_DIM + colBase, H_DIM, &smem[0][OFF_BEW], 32, 64, tid, 256);
    copy_tile_bf16(wwx + (size_t)k * H_DIM + colBase, H_DIM, &smem[0][OFF_BWX], 32, 64, tid, 256);
    __syncthreads();
    compute_step(&smem[0][0], waveM, waveN, lane, accT, accE, accX);
  }
#endif

  // Epilogue: h = sigmoid(T+bT)*tanh(state) + sigmoid(E+bE)*tanh(X); write twice.
#pragma unroll
  for (int mi = 0; mi < 2; ++mi) {
#pragma unroll
    for (int ni = 0; ni < 2; ++ni) {
#pragma unroll
      for (int v = 0; v < 8; ++v) {
        const int row = rowBase + waveM * 32 + mi * 16 + ((lane >> 4) * 8 + v);
        const int col = colBase + waveN * 32 + ni * 16 + (lane & 15);
        const size_t o = (size_t)row * H_DIM + col;
        const float t = sigmoidf_(accT[mi][ni][v] + theta_b[col]);
        const float e = sigmoidf_(accE[mi][ni][v] + eta_b[col]);
        const float h = t * tanhf(state_f32[o]) + e * tanhf(accX[mi][ni][v]);
        out[o] = h;
        out[(size_t)B_DIM * H_DIM + o] = h;
      }
    }
  }
}

// ---------------------------------------------------------------------------
// Fallback (ws too small): round-1 fused kernel with in-loop f32->bf16 staging.
// ---------------------------------------------------------------------------
__device__ __forceinline__ void g2s_bf16_tile(const float* __restrict__ g, int ldg,
                                              __bf16* __restrict__ s,
                                              int rows, int cols, int tid, int nthr) {
  const int total4 = (rows * cols) >> 2;
  for (int i = tid; i < total4; i += nthr) {
    const int idx = i << 2;
    const int r = idx / cols;
    const int c = idx - r * cols;
    const float4 v = *(const float4*)(g + (size_t)r * ldg + c);
    union { __bf16 h[4]; uint2 u; } p;
    p.h[0] = (__bf16)v.x; p.h[1] = (__bf16)v.y;
    p.h[2] = (__bf16)v.z; p.h[3] = (__bf16)v.w;
    *(uint2*)(s + idx) = p.u;
  }
}

__global__ __launch_bounds__(256)
void cfn_cell_fused_kernel(const float* __restrict__ inputs,
                           const float* __restrict__ state,
                           const float* __restrict__ theta_u_w,
                           const float* __restrict__ theta_w_w,
                           const float* __restrict__ theta_w_b,
                           const float* __restrict__ eta_u_w,
                           const float* __restrict__ eta_w_w,
                           const float* __restrict__ eta_w_b,
                           const float* __restrict__ wx_w,
                           float* __restrict__ out) {
  __shared__ __bf16 smem[BUF_ELEMS];

  const int tid   = threadIdx.x;
  const int lane  = tid & 31;
  const int wave  = tid >> 5;
  const int waveM = wave & 3;
  const int waveN = wave >> 2;
  const int rowBase = blockIdx.y * 128;
  const int colBase = blockIdx.x * 64;

  v8f accT[2][2] = {};
  v8f accE[2][2] = {};
  v8f accX[2][2] = {};

  for (int k = 0; k < H_DIM; k += 32) {
    __syncthreads();
    g2s_bf16_tile(state     + (size_t)rowBase * H_DIM + k, H_DIM, &smem[OFF_AS], 128, 32, tid, 256);
    g2s_bf16_tile(inputs    + (size_t)rowBase * DIN   + k, DIN,   &smem[OFF_AI], 128, 32, tid, 256);
    g2s_bf16_tile(theta_u_w + (size_t)k * H_DIM + colBase, H_DIM, &smem[OFF_BTU], 32, 64, tid, 256);
    g2s_bf16_tile(theta_w_w + (size_t)k * H_DIM + colBase, H_DIM, &smem[OFF_BTW], 32, 64, tid, 256);
    g2s_bf16_tile(eta_u_w   + (size_t)k * H_DIM + colBase, H_DIM, &smem[OFF_BEU], 32, 64, tid, 256);
    g2s_bf16_tile(eta_w_w   + (size_t)k * H_DIM + colBase, H_DIM, &smem[OFF_BEW], 32, 64, tid, 256);
    g2s_bf16_tile(wx_w      + (size_t)k * H_DIM + colBase, H_DIM, &smem[OFF_BWX], 32, 64, tid, 256);
    __syncthreads();
    compute_step(smem, waveM, waveN, lane, accT, accE, accX);
  }

#pragma unroll
  for (int mi = 0; mi < 2; ++mi) {
#pragma unroll
    for (int ni = 0; ni < 2; ++ni) {
#pragma unroll
      for (int v = 0; v < 8; ++v) {
        const int row = rowBase + waveM * 32 + mi * 16 + ((lane >> 4) * 8 + v);
        const int col = colBase + waveN * 32 + ni * 16 + (lane & 15);
        const size_t o = (size_t)row * H_DIM + col;
        const float t = sigmoidf_(accT[mi][ni][v] + theta_w_b[col]);
        const float e = sigmoidf_(accE[mi][ni][v] + eta_w_b[col]);
        const float h = t * tanhf(state[o]) + e * tanhf(accX[mi][ni][v]);
        out[o] = h;
        out[(size_t)B_DIM * H_DIM + o] = h;
      }
    }
  }
}

extern "C" void kernel_launch(void* const* d_in, const int* in_sizes, int n_in,
                              void* d_out, int out_size, void* d_ws, size_t ws_size,
                              hipStream_t stream) {
  const float* inputs    = (const float*)d_in[0];
  const float* state     = (const float*)d_in[1];
  const float* theta_u_w = (const float*)d_in[2];
  const float* theta_w_w = (const float*)d_in[3];
  const float* theta_w_b = (const float*)d_in[4];
  const float* eta_u_w   = (const float*)d_in[5];
  const float* eta_w_w   = (const float*)d_in[6];
  const float* eta_w_b   = (const float*)d_in[7];
  const float* wx_w      = (const float*)d_in[8];
  float* out = (float*)d_out;

  const size_t nIn = (size_t)B_DIM * DIN;
  const size_t nSt = (size_t)B_DIM * H_DIM;
  const size_t nW  = (size_t)H_DIM * H_DIM;
  const size_t needBytes = (nIn + nSt + 5 * nW) * 2;  // bf16 pack of all operands

  dim3 grid(H_DIM / 64, B_DIM / 128);  // (32, 32)

  if (ws_size >= needBytes) {
    __bf16* p = (__bf16*)d_ws;
    __bf16* inputs_bf = p;  p += nIn;
    __bf16* state_bf  = p;  p += nSt;
    __bf16* wtu_bf = p;     p += nW;
    __bf16* wtw_bf = p;     p += nW;
    __bf16* weu_bf = p;     p += nW;
    __bf16* wew_bf = p;     p += nW;
    __bf16* wwx_bf = p;

    auto cvt = [&](const float* s, __bf16* d, size_t n) {
      cvt_bf16_kernel<<<dim3((unsigned)(n / (8 * 256))), dim3(256), 0, stream>>>(s, d, (int)n);
    };
    cvt(inputs,    inputs_bf, nIn);
    cvt(state,     state_bf,  nSt);
    cvt(theta_u_w, wtu_bf, nW);
    cvt(theta_w_w, wtw_bf, nW);
    cvt(eta_u_w,   weu_bf, nW);
    cvt(eta_w_w,   wew_bf, nW);
    cvt(wx_w,      wwx_bf, nW);

    cfn_gemm_bf16_kernel<<<grid, dim3(256), 0, stream>>>(
        inputs_bf, state_bf, wtu_bf, wtw_bf, weu_bf, wew_bf, wwx_bf,
        state, theta_w_b, eta_w_b, out);
  } else {
    cfn_cell_fused_kernel<<<grid, dim3(256), 0, stream>>>(
        inputs, state, theta_u_w, theta_w_w, theta_w_b,
        eta_u_w, eta_w_w, eta_w_b, wx_w, out);
  }
}
